// _MambaMixer_240518168877
// MI455X (gfx1250) — matrix operand order repaired and checked by reading, not compiled, not run on hardware
//
#include <hip/hip_runtime.h>
#include <hip/hip_bf16.h>

// Problem constants (from reference): B=2, L=1024, D=1024, E=2 -> DI=2048,
// S=16, K(conv)=4, R=64, proj width = R+2S = 96.
#define BB    2
#define LL    1024
#define DD    1024
#define DI_   2048
#define SS    16
#define RR    64
#define KC    4
#define NPROJ 96
#define M_TOT (BB * LL)   // 2048 flattened (b,l) rows

typedef __attribute__((ext_vector_type(16))) _Float16 v16h;
typedef __attribute__((ext_vector_type(8)))  _Float16 v8h;
typedef __attribute__((ext_vector_type(8)))  float    v8f;

// ---------------------------------------------------------------------------
// f16 fragment loaders (operands pre-converted to f16 in workspace).
// A: 16x32 (MxK) row-major, lane = hi*16+m holds K in
//    {16*hi..16*hi+15}  -> two 16B loads.
// B: 32x16 (KxN) == W^T, W row-major (N,K); lane = hi*16+n holds
//    K in {16*hi..16*hi+15} -> one 32B contiguous load.
// ---------------------------------------------------------------------------
__device__ inline v16h load_a16(const _Float16* __restrict__ A, int lda,
                                int m0, int k0, int lane) {
  const int m  = m0 + (lane & 15);
  const int hi = lane >> 4;
  const _Float16* p = A + (size_t)m * lda + k0 + 16 * hi;
  const v8h c0 = *(const v8h*)p;
  const v8h c1 = *(const v8h*)(p + 8);
  return __builtin_shufflevector(c0, c1, 0, 1, 2, 3, 4, 5, 6, 7,
                                 8, 9, 10, 11, 12, 13, 14, 15);
}

__device__ inline v16h load_b16(const _Float16* __restrict__ W, int ldb,
                                int n0, int k0, int lane) {
  const int n  = n0 + (lane & 15);
  const int hi = lane >> 4;
  return *(const v16h*)(W + (size_t)n * ldb + k0 + 16 * hi);
}

// ---------------------------------------------------------------------------
// WMMA GEMM: C[M,N] = A[M,K](f16) * W[N,K](f16)^T (+bias) (+epilogue)
// Wave tile: (16*MT) x (16*NT).  Grid sized exactly (waves == tiles).
// MODE 0: f32 store; MODE 1: softplus f32 store; MODE 2: f32 + f16 dual store.
// ---------------------------------------------------------------------------
template <int MT, int NT, int MODE>
__global__ __launch_bounds__(128) void wmma_gemm_h(
    const _Float16* __restrict__ A, const _Float16* __restrict__ W,
    const float* __restrict__ bias, float* __restrict__ C,
    _Float16* __restrict__ C16, int M, int N, int K, int lda, int ldb,
    int ldc) {
  const int lane   = threadIdx.x & 31;
  const int wid    = blockIdx.x * (blockDim.x >> 5) + (threadIdx.x >> 5);
  const int tilesN = N / (16 * NT);
  const int m0     = (wid / tilesN) * (16 * MT);
  const int n0     = (wid % tilesN) * (16 * NT);
  if (m0 >= M) return;  // wave-uniform (EXEC stays all-ones for WMMA)

  v8f acc[MT][NT];
#pragma unroll
  for (int i = 0; i < MT; ++i)
#pragma unroll
    for (int t = 0; t < NT; ++t) {
      v8f z = {};
      acc[i][t] = z;
    }

  for (int kk = 0; kk < K; kk += 32) {
    v16h a[MT];
#pragma unroll
    for (int i = 0; i < MT; ++i) a[i] = load_a16(A, lda, m0 + 16 * i, kk, lane);
#pragma unroll
    for (int t = 0; t < NT; ++t) {
      const v16h b = load_b16(W, ldb, n0 + 16 * t, kk, lane);
#pragma unroll
      for (int i = 0; i < MT; ++i) {
        acc[i][t] = __builtin_amdgcn_wmma_f32_16x16x32_f16(
            false, a[i], false, b, (short)0, acc[i][t], false, false);
      }
    }
  }

  const int hi    = lane >> 4;
  const int nlane = lane & 15;
#pragma unroll
  for (int i = 0; i < MT; ++i) {
#pragma unroll
    for (int t = 0; t < NT; ++t) {
      const int n  = n0 + 16 * t + nlane;
      const float bv = bias ? bias[n] : 0.0f;
#pragma unroll
      for (int r = 0; r < 8; ++r) {
        const size_t row = (size_t)(m0 + 16 * i + r + 8 * hi);
        float v = acc[i][t][r] + bv;
        if (MODE == 1) v = (v > 20.0f) ? v : __logf(1.0f + __expf(v));
        C[row * ldc + n] = v;
        if (MODE == 2) C16[row * ldc + n] = (_Float16)v;
      }
    }
  }
}

// ---------------------------------------------------------------------------
// f32 -> f16 bulk conversion (8 elements / thread, b128 in, b128 out).
// ---------------------------------------------------------------------------
__global__ __launch_bounds__(256) void cvt16_kernel(
    const float* __restrict__ s, _Float16* __restrict__ d) {
  const size_t i = ((size_t)blockIdx.x * 256 + threadIdx.x) * 8;
  const float4 x0 = *(const float4*)(s + i);
  const float4 x1 = *(const float4*)(s + i + 4);
  v8h o;
  o[0] = (_Float16)x0.x; o[1] = (_Float16)x0.y;
  o[2] = (_Float16)x0.z; o[3] = (_Float16)x0.w;
  o[4] = (_Float16)x1.x; o[5] = (_Float16)x1.y;
  o[6] = (_Float16)x1.z; o[7] = (_Float16)x1.w;
  *(v8h*)(d + i) = o;
}

// ---------------------------------------------------------------------------
// Depthwise causal conv (K=4) + bias + SiLU.
// x = xz[:, :, 0:DI] (xz row-major (b*L, 2*DI)); writes u in f32 (for scan)
// and f16 (GEMM operand), both (B, L, DI) row-major.
// ---------------------------------------------------------------------------
__global__ __launch_bounds__(256) void conv_silu_kernel(
    const float* __restrict__ xz, const float* __restrict__ Wc,
    const float* __restrict__ bc, float* __restrict__ u,
    _Float16* __restrict__ u16) {
  const int idx = blockIdx.x * 256 + threadIdx.x;  // (b*L + l)*DI + c
  const int c   = idx & (DI_ - 1);
  const int bl  = idx >> 11;                       // b*L + l
  const int l   = bl & (LL - 1);
  float acc = bc[c];
  const float* w = Wc + c * KC;
#pragma unroll
  for (int j = 0; j < KC; ++j) {
    const int ll = l + j - (KC - 1);
    if (ll >= 0) {
      acc += xz[(size_t)(bl + j - (KC - 1)) * (2 * DI_) + c] * w[j];
    }
  }
  const float s = acc / (1.0f + __expf(-acc));  // SiLU
  u[(size_t)idx]   = s;
  u16[(size_t)idx] = (_Float16)s;
}

// ---------------------------------------------------------------------------
// Selective scan + output gating.
// One thread per (b, channel): 16-state linear recurrence over L steps.
// Bg/Cg (proj cols 64..95) shared per timestep -> staged through LDS.
// gated16[b,t,c] = (sum_s C_t[s]*h_t[s]) * silu(z[b,t,c])   (f16, GEMM operand)
// ---------------------------------------------------------------------------
#define TCH 64
__global__ __launch_bounds__(256) void scan_gate_kernel(
    const float* __restrict__ u, const float* __restrict__ dt,
    const float* __restrict__ proj, const float* __restrict__ A_log,
    const float* __restrict__ xz, _Float16* __restrict__ gated16) {
  __shared__ float lB[TCH * SS];
  __shared__ float lC[TCH * SS];
  const int b = blockIdx.y;
  const int c = blockIdx.x * 256 + threadIdx.x;

  float Arow[SS];
#pragma unroll
  for (int s = 0; s < SS; ++s) Arow[s] = -__expf(A_log[c * SS + s]);
  float h[SS];
#pragma unroll
  for (int s = 0; s < SS; ++s) h[s] = 0.0f;

  for (int t0 = 0; t0 < LL; t0 += TCH) {
    __syncthreads();
    for (int i = threadIdx.x; i < TCH * SS; i += 256) {
      const int tt = i >> 4;
      const int s  = i & 15;
      const float* pr = proj + (size_t)(b * LL + t0 + tt) * NPROJ;
      lB[i] = pr[RR + s];
      lC[i] = pr[RR + SS + s];
    }
    __syncthreads();

    for (int tt = 0; tt < TCH; ++tt) {
      const size_t row = (size_t)(b * LL + t0 + tt);
      const float dtv = dt[row * DI_ + c];
      const float du  = dtv * u[row * DI_ + c];
      const float* Bt = lB + tt * SS;
      const float* Ct = lC + tt * SS;
      float y = 0.0f;
#pragma unroll
      for (int s = 0; s < SS; ++s) {
        h[s] = __expf(dtv * Arow[s]) * h[s] + du * Bt[s];
        y += h[s] * Ct[s];
      }
      const float z = xz[row * (2 * DI_) + DI_ + c];
      gated16[row * DI_ + c] = (_Float16)(y * (z / (1.0f + __expf(-z))));
    }
  }
}

// ---------------------------------------------------------------------------
// Host-side pipeline.
// ---------------------------------------------------------------------------
extern "C" void kernel_launch(void* const* d_in, const int* in_sizes, int n_in,
                              void* d_out, int out_size, void* d_ws,
                              size_t ws_size, hipStream_t stream) {
  (void)in_sizes; (void)n_in; (void)out_size; (void)ws_size;
  const float* inputs = (const float*)d_in[0];
  const float* W_in   = (const float*)d_in[1];
  const float* W_conv = (const float*)d_in[2];
  const float* b_conv = (const float*)d_in[3];
  const float* W_x    = (const float*)d_in[4];
  const float* W_dt   = (const float*)d_in[5];
  const float* b_dt   = (const float*)d_in[6];
  const float* A_log  = (const float*)d_in[7];
  const float* W_out  = (const float*)d_in[8];
  float* out = (float*)d_out;

  // Workspace carve-up (256B-aligned regions).
  char* wp = (char*)d_ws;
  auto alloc = [&](size_t bytes) -> void* {
    void* p = (void*)wp;
    wp += (bytes + 255) & ~(size_t)255;
    return p;
  };
  float*     xz      = (float*)alloc((size_t)M_TOT * 2 * DI_ * 4);   // 32 MB
  float*     feats   = (float*)alloc((size_t)M_TOT * DI_ * 4);       // 16 MB
  float*     proj    = (float*)alloc((size_t)M_TOT * NPROJ * 4);     // .75 MB
  float*     dtb     = (float*)alloc((size_t)M_TOT * DI_ * 4);       // 16 MB
  _Float16*  feats16 = (_Float16*)alloc((size_t)M_TOT * DI_ * 2);    //  8 MB
  _Float16*  proj16  = (_Float16*)alloc((size_t)M_TOT * NPROJ * 2);  // .4 MB
  _Float16*  gated16 = (_Float16*)alloc((size_t)M_TOT * DI_ * 2);    //  8 MB
  _Float16*  in16    = (_Float16*)alloc((size_t)M_TOT * DD * 2);     //  4 MB
  _Float16*  Win16   = (_Float16*)alloc((size_t)2 * DI_ * DD * 2);   //  8 MB
  _Float16*  Wx16    = (_Float16*)alloc((size_t)NPROJ * DI_ * 2);    // .4 MB
  _Float16*  Wdt16   = (_Float16*)alloc((size_t)DI_ * RR * 2);       // .25 MB
  _Float16*  Wout16  = (_Float16*)alloc((size_t)DD * DI_ * 2);       //  4 MB

  // 0) one-shot f32 -> f16 staging of GEMM operands (all counts % 2048 == 0)
  cvt16_kernel<<<(M_TOT * DD) / 2048, 256, 0, stream>>>(inputs, in16);
  cvt16_kernel<<<(2 * DI_ * DD) / 2048, 256, 0, stream>>>(W_in, Win16);
  cvt16_kernel<<<(NPROJ * DI_) / 2048, 256, 0, stream>>>(W_x, Wx16);
  cvt16_kernel<<<(DI_ * RR) / 2048, 256, 0, stream>>>(W_dt, Wdt16);
  cvt16_kernel<<<(DD * DI_) / 2048, 256, 0, stream>>>(W_out, Wout16);

  const int WPB = 4;  // waves per block (128 threads)

  // 1) xz = inputs @ W_in^T : M=2048, N=4096, K=1024; 32x64 wave tiles
  {
    const int tiles = (M_TOT / 32) * ((2 * DI_) / 64);  // 4096 waves
    wmma_gemm_h<2, 4, 0><<<tiles / WPB, 128, 0, stream>>>(
        in16, Win16, nullptr, xz, nullptr, M_TOT, 2 * DI_, DD, DD, DD,
        2 * DI_);
  }

  // 2) depthwise conv + SiLU -> feats (f32) + feats16
  conv_silu_kernel<<<(M_TOT * DI_) / 256, 256, 0, stream>>>(
      xz, W_conv, b_conv, feats, feats16);

  // 3) proj = feats @ W_x^T : M=2048, N=96, K=2048; 32x48 tiles, dual store
  {
    const int tiles = (M_TOT / 32) * (NPROJ / 48);  // 128 waves
    wmma_gemm_h<2, 3, 2><<<tiles / WPB, 128, 0, stream>>>(
        feats16, Wx16, nullptr, proj, proj16, M_TOT, NPROJ, DI_, DI_, DI_,
        NPROJ);
  }

  // 4) dt = softplus(proj[:, :64] @ W_dt^T + b_dt) : M=2048, N=2048, K=64
  {
    const int tiles = (M_TOT / 32) * (DI_ / 64);  // 2048 waves
    wmma_gemm_h<2, 4, 1><<<tiles / WPB, 128, 0, stream>>>(
        proj16, Wdt16, b_dt, dtb, nullptr, M_TOT, DI_, RR, NPROJ, RR, DI_);
  }

  // 5) selective scan + gating -> gated16 (B, L, DI)
  {
    dim3 grid(DI_ / 256, BB);
    scan_gate_kernel<<<grid, 256, 0, stream>>>(feats, dtb, proj, A_log, xz,
                                               gated16);
  }

  // 6) out = gated @ W_out^T : M=2048, N=1024, K=2048
  {
    const int tiles = (M_TOT / 32) * (DD / 64);  // 1024 waves
    wmma_gemm_h<2, 4, 0><<<tiles / WPB, 128, 0, stream>>>(
        gated16, Wout16, nullptr, out, nullptr, M_TOT, DD, DI_, DI_, DI_, DD);
  }
}
